// PairwiseLoss_24696061952425
// MI455X (gfx1250) — matrix operand order repaired and checked by reading, not compiled, not run on hardware
//
#include <hip/hip_runtime.h>
#include <hip/hip_bf16.h>

typedef __attribute__((ext_vector_type(16))) _Float16 v16h;
typedef __attribute__((ext_vector_type(8)))  _Float16 v8h;
typedef __attribute__((ext_vector_type(8)))  float    v8f;
typedef __attribute__((ext_vector_type(4)))  float    v4f;

#define NKEEP 900      // feature length
#define NPAD  928      // f16 row padding (29 chunks of 32, 16B-aligned rows)
#define CROWS 32768    // cache rows
#define BATCH 128      // batch
#define NVALS 901      // possible count values 0..900
#define INVALID 0xFFFFFFFFu

// ---------------------------------------------------------------------------
// Kernel 0: convert y_hat / y_true to zero-padded f16 rows [BATCH][NPAD]
// ---------------------------------------------------------------------------
__global__ void pl_cvt_y(const float* __restrict__ yh, const float* __restrict__ yt,
                         _Float16* __restrict__ yh16, _Float16* __restrict__ yt16) {
  int i = blockIdx.x * blockDim.x + threadIdx.x;
  if (i >= BATCH * NPAD) return;
  int b = i / NPAD, n = i % NPAD;
  float vh = 0.0f, vt = 0.0f;
  if (n < NKEEP) { vh = yh[b * NKEEP + n]; vt = yt[b * NKEEP + n]; }
  yh16[i] = (_Float16)vh;
  yt16[i] = (_Float16)vt;
}

// ---------------------------------------------------------------------------
// Kernel 1: init minidx table
// ---------------------------------------------------------------------------
__global__ void pl_init(unsigned* __restrict__ minidx, int n) {
  int i = blockIdx.x * blockDim.x + threadIdx.x;
  if (i < n) minidx[i] = INVALID;
}

// ---------------------------------------------------------------------------
// Fragment loaders (layouts per ISA 7.12.2)
// ---------------------------------------------------------------------------
// A 16x32 f16: lane l holds row M=l&15; elems 0..7 -> K=na+0..7, 8..15 -> K=na+16..23
__device__ inline v16h load_a16(const _Float16* __restrict__ row, int na) {
  v8h lo = *(const v8h*)(row + na);        // 16B aligned (na multiple of 8)
  v8h hi = *(const v8h*)(row + na + 16);
  return __builtin_shufflevector(lo, hi, 0,1,2,3,4,5,6,7,8,9,10,11,12,13,14,15);
}

// B 32x16: lane l holds column N=l&15; elems 0..7 -> K=nb+0..7, 8..15 -> K=nb+16..23 (f32 -> f16)
__device__ inline v16h load_b_full(const float* __restrict__ row, int nb) {
  v16h f;
#pragma unroll
  for (int i = 0; i < 16; ++i) f[i] = (_Float16)row[nb + (i < 8 ? i : i + 8)];
  return f;
}

// ---------------------------------------------------------------------------
// Kernel 2: fused dual GEMM (counts + scores) via WMMA f16->f32
// Block = 128 threads = 4 waves; wave w: cache rows [bx*64+w*16,+16) x batches
// [by*16,+16). Both matmuls share one pass over cache.
// ---------------------------------------------------------------------------
__global__ __launch_bounds__(128)
void pl_gemm(const _Float16* __restrict__ yh16,
             const _Float16* __restrict__ yt16,
             const float*    __restrict__ cache,
             float*          __restrict__ S,        // [BATCH][CROWS]
             unsigned*       __restrict__ minidx)   // [BATCH][NVALS]
{
  const int lane    = threadIdx.x & 31;
  const int wv      = threadIdx.x >> 5;
  const int halfsel = lane >> 4;
  const int lrow    = lane & 15;

  const int kTile = blockIdx.x * 64 + wv * 16;  // cache-row tile (N dim)
  const int bTile = blockIdx.y * 16;            // batch tile     (M dim)

  const float*    cacheRow = cache + (size_t)(kTile + lrow) * NKEEP;
  const _Float16* ytRow    = yt16  + (size_t)(bTile + lrow) * NPAD;
  const _Float16* yhRow    = yh16  + (size_t)(bTile + lrow) * NPAD;

  v8f accC = {};  // counts accumulator (exact: 0/1 f16 products, f32 adds)
  v8f accS = {};  // scores accumulator

  // 28 full K-chunks (n = 0..895), all loads unguarded
  for (int kk = 0; kk < 896; kk += 32) {
    v16h bf = load_b_full(cacheRow, kk + halfsel * 8);
    v16h at = load_a16(ytRow, kk + halfsel * 8);
    v16h ah = load_a16(yhRow, kk + halfsel * 8);
    accC = __builtin_amdgcn_wmma_f32_16x16x32_f16(false, at, false, bf,
                                                  (short)0, accC, false, false);
    accS = __builtin_amdgcn_wmma_f32_16x16x32_f16(false, ah, false, bf,
                                                  (short)0, accS, false, false);
  }
  // Tail chunk: only n = 896..899 carry data. A-side is zero-padded to 928;
  // B-side: one in-bounds aligned b128 load (bytes 3584..3599 of each row),
  // lanes with halfsel==1 contribute zeros (exec-masked VALU only, no loads).
  {
    v4f t = *(const v4f*)(cacheRow + 896);
    v16h bf = {};
    if (halfsel == 0) {
      bf[0] = (_Float16)t.x; bf[1] = (_Float16)t.y;
      bf[2] = (_Float16)t.z; bf[3] = (_Float16)t.w;
    }
    v16h at = load_a16(ytRow, 896 + halfsel * 8);
    v16h ah = load_a16(yhRow, 896 + halfsel * 8);
    accC = __builtin_amdgcn_wmma_f32_16x16x32_f16(false, at, false, bf,
                                                  (short)0, accC, false, false);
    accS = __builtin_amdgcn_wmma_f32_16x16x32_f16(false, ah, false, bf,
                                                  (short)0, accS, false, false);
  }

  // Epilogue. D layout: VGPR r, lane l -> M = r + 8*(l>>4), N = l&15.
  const int k = kTile + lrow;
#pragma unroll
  for (int r = 0; r < 8; ++r) {
    const int brow = bTile + r + 8 * halfsel;
    S[(size_t)brow * CROWS + k] = accS[r];
    int cnt = (int)rintf(accC[r]);
    cnt = min(max(cnt, 0), NKEEP);
    atomicMin(&minidx[brow * NVALS + cnt], (unsigned)k);
  }
}

// ---------------------------------------------------------------------------
// Kernel 3: per-batch loss. One block per batch item.
// ---------------------------------------------------------------------------
__global__ __launch_bounds__(256)
void pl_finalize(const unsigned* __restrict__ minidx,
                 const float* __restrict__ S,
                 float* __restrict__ losses)
{
  const int b   = blockIdx.x;
  const int tid = threadIdx.x;
  const unsigned* row = minidx + b * NVALS;
  const float*   Srow = S + (size_t)b * CROWS;

  __shared__ int   s_max[256];
  __shared__ float s_sum[256];
  __shared__ float s_cnt[256];
  __shared__ int   s_dmax;
  __shared__ float s_best;

  // dmax = largest count value present (first occurrence of min v after sort)
  int localmax = -1;
  for (int d = tid; d < NVALS; d += 256)
    if (row[d] != INVALID) localmax = d;   // d increases -> keeps the max
  s_max[tid] = localmax;
  __syncthreads();
  for (int off = 128; off > 0; off >>= 1) {
    if (tid < off) s_max[tid] = max(s_max[tid], s_max[tid + off]);
    __syncthreads();
  }
  if (tid == 0) {
    s_dmax = s_max[0];
    s_best = Srow[row[s_max[0]]];
  }
  __syncthreads();
  const int   dmax = s_dmax;
  const float best = s_best;

  // sum relu(s[first_occurrence(d)] - best) over distinct d != dmax
  float sum = 0.0f, cnt = 0.0f;
  for (int d = tid; d < NVALS; d += 256) {
    unsigned idx = row[d];
    if (idx != INVALID && d != dmax) {
      float t = Srow[idx] - best;
      sum += (t > 0.0f) ? t : 0.0f;
      cnt += 1.0f;
    }
  }
  s_sum[tid] = sum; s_cnt[tid] = cnt;
  __syncthreads();
  for (int off = 128; off > 0; off >>= 1) {
    if (tid < off) { s_sum[tid] += s_sum[tid + off]; s_cnt[tid] += s_cnt[tid + off]; }
    __syncthreads();
  }
  if (tid == 0) losses[b] = s_sum[0] / fmaxf(s_cnt[0], 1.0f);
}

// ---------------------------------------------------------------------------
// Kernel 4: deterministic ordered mean over the batch
// ---------------------------------------------------------------------------
__global__ void pl_mean(const float* __restrict__ losses, float* __restrict__ out) {
  if (threadIdx.x == 0 && blockIdx.x == 0) {
    float s = 0.0f;
    for (int i = 0; i < BATCH; ++i) s += losses[i];
    out[0] = s / (float)BATCH;
  }
}

// ---------------------------------------------------------------------------
extern "C" void kernel_launch(void* const* d_in, const int* in_sizes, int n_in,
                              void* d_out, int out_size, void* d_ws, size_t ws_size,
                              hipStream_t stream) {
  const float* y_hat  = (const float*)d_in[0];   // [128, 900]
  const float* y_true = (const float*)d_in[1];   // [128, 900]
  // d_in[2] = sol_true (unused by the reference loss)
  const float* cache  = (const float*)d_in[3];   // [32768, 900]
  float* out = (float*)d_out;

  // workspace carve-up
  char* ws = (char*)d_ws;
  size_t off = 0;
  float*    S      = (float*)(ws + off);  off += (size_t)BATCH * CROWS * 4;  // 16.78 MB
  unsigned* minidx = (unsigned*)(ws + off); off += (size_t)BATCH * NVALS * 4; // 461 KB
  float*    losses = (float*)(ws + off);  off += BATCH * 4;
  _Float16* yh16   = (_Float16*)(ws + off); off += (size_t)BATCH * NPAD * 2;  // 232 KB
  _Float16* yt16   = (_Float16*)(ws + off);

  const int ncvt = BATCH * NPAD;
  pl_cvt_y<<<(ncvt + 255) / 256, 256, 0, stream>>>(y_hat, y_true, yh16, yt16);

  const int ninit = BATCH * NVALS;
  pl_init<<<(ninit + 255) / 256, 256, 0, stream>>>(minidx, ninit);

  dim3 grid(CROWS / 64, BATCH / 16);
  pl_gemm<<<grid, 128, 0, stream>>>(yh16, yt16, cache, S, minidx);

  pl_finalize<<<BATCH, 256, 0, stream>>>(minidx, S, losses);
  pl_mean<<<1, 32, 0, stream>>>(losses, out);
}
